// TransformerDecoder_46780783788637
// MI455X (gfx1250) — matrix operand order repaired and checked by reading, not compiled, not run on hardware
//
#include <hip/hip_runtime.h>
#include <stdint.h>

#define D_    128
#define H_    8
#define T_    2048
#define B_    2
#define DEPTH_ 6
#define HD_   1024          // H_*D_
#define BT_   4096          // B_*T_
#define C_    32000

typedef __attribute__((ext_vector_type(16))) __bf16   v16bf;
typedef __attribute__((ext_vector_type(8)))  float    v8f;
typedef __attribute__((ext_vector_type(4)))  unsigned int u32x4;

union BFrag { v16bf v; u32x4 q[2]; };

__device__ __forceinline__ uint16_t f2bf(float f) {
  union { float f; uint32_t u; } c; c.f = f;
  uint32_t u = c.u;
  u += 0x7FFFu + ((u >> 16) & 1u);          // round-to-nearest-even
  return (uint16_t)(u >> 16);
}

__device__ __forceinline__ v8f vzero8() {
  v8f z = {0.f, 0.f, 0.f, 0.f, 0.f, 0.f, 0.f, 0.f};
  return z;
}

// ---------------- embedding: hid = tok_emb[x] + pos_emb ----------------
__global__ __launch_bounds__(128) void embed_kernel(
    const int* __restrict__ x, const float* __restrict__ te,
    const float* __restrict__ pe, float* __restrict__ hid,
    uint16_t* __restrict__ hidb)
{
  int tok = blockIdx.x, i = threadIdx.x;
  int t = tok & (T_ - 1);
  int id = x[tok];
  float v = te[(size_t)id * D_ + i] + pe[(size_t)t * D_ + i];
  hid[(size_t)tok * D_ + i]  = v;
  hidb[(size_t)tok * D_ + i] = f2bf(v);
}

// ------------- weight transpose+convert: fp32[K][N] -> bf16[N][K] -------------
__global__ __launch_bounds__(256) void transw_kernel(
    const float* __restrict__ in, uint16_t* __restrict__ out, int K, int N)
{
  int idx = blockIdx.x * 256 + threadIdx.x;     // idx = n*K + k, total = K*N (mult of 256)
  int n = idx / K, k = idx - n * K;
  out[idx] = f2bf(in[(size_t)k * N + n]);
}

// ------------- V transpose: vb[b][t][h][d] -> vT[b][h][d][t] (bf16) -------------
__global__ __launch_bounds__(256) void transv_kernel(
    const uint16_t* __restrict__ vb, uint16_t* __restrict__ vT)
{
  size_t idx = (size_t)blockIdx.x * 256 + threadIdx.x;  // total = B*H*D*T
  int t = (int)(idx & (T_ - 1));
  size_t r = idx >> 11;
  int d = (int)(r & (D_ - 1)); r >>= 7;
  int h = (int)(r & (H_ - 1));
  int b = (int)(r >> 3);
  vT[idx] = vb[(((size_t)b * T_ + t) * H_ + h) * D_ + d];
}

// ---------------- bf16 GEMM: C[M,N] = A[M,K] * B^T[N,K] (+bias)(relu) ----------------
__global__ __launch_bounds__(128) void gemm_kernel(
    const uint16_t* __restrict__ A, const uint16_t* __restrict__ BT,
    const float* __restrict__ bias, float* __restrict__ outF,
    uint16_t* __restrict__ outB, int M, int N, int K, int relu)
{
  int lane = threadIdx.x & 31;
  int wave = threadIdx.x >> 5;
  int hl = lane >> 4, l16 = lane & 15;
  int mB = blockIdx.y * 64 + (wave >> 1) * 32;
  int nB = blockIdx.x * 64 + (wave & 1) * 32;

  v8f acc[2][2];
#pragma unroll
  for (int i = 0; i < 2; ++i)
#pragma unroll
    for (int j = 0; j < 2; ++j) acc[i][j] = vzero8();

  for (int kb = 0; kb < K; kb += 32) {
    BFrag a[2], b[2];
#pragma unroll
    for (int i = 0; i < 2; ++i) {
      const uint16_t* ap = A + (size_t)(mB + i * 16 + l16) * K + kb + 8 * hl;
      a[i].q[0] = *(const u32x4*)ap;            // K: kb+8h .. kb+8h+7
      a[i].q[1] = *(const u32x4*)(ap + 16);     // K: kb+16+8h .. kb+23+8h
      __builtin_prefetch(ap + 32, 0, 1);        // global_prefetch_b8, next K tile
    }
#pragma unroll
    for (int j = 0; j < 2; ++j) {
      const uint16_t* bp = BT + (size_t)(nB + j * 16 + l16) * K + kb + 8 * hl;
      b[j].q[0] = *(const u32x4*)bp;            // K: kb+8h .. kb+8h+7
      b[j].q[1] = *(const u32x4*)(bp + 16);     // K: kb+16+8h .. kb+23+8h
    }
#pragma unroll
    for (int i = 0; i < 2; ++i)
#pragma unroll
      for (int j = 0; j < 2; ++j)
        acc[i][j] = __builtin_amdgcn_wmma_f32_16x16x32_bf16(
            false, a[i].v, false, b[j].v, (short)0, acc[i][j], false, false);
  }

#pragma unroll
  for (int i = 0; i < 2; ++i)
#pragma unroll
    for (int j = 0; j < 2; ++j) {
      int col = nB + j * 16 + l16;
      float bv = bias ? bias[col] : 0.0f;
#pragma unroll
      for (int r = 0; r < 8; ++r) {
        int row = mB + i * 16 + 8 * hl + r;
        float v = acc[i][j][r] + bv;
        if (relu) v = fmaxf(v, 0.0f);
        size_t o = (size_t)row * N + col;
        if (outF) outF[o] = v;
        if (outB) outB[o] = f2bf(v);
      }
    }
}

// ---------------- flash attention (causal), bf16 WMMA, fp32 accumulate ----------------
__global__ __launch_bounds__(128) void attn_kernel(
    const uint16_t* __restrict__ qb, const uint16_t* __restrict__ kbuf,
    const uint16_t* __restrict__ vT, uint16_t* __restrict__ attnb)
{
  __shared__ __align__(16) uint16_t Pst[4][16 * 32];
  int lane = threadIdx.x & 31;
  int wave = threadIdx.x >> 5;
  int hl = lane >> 4, l16 = lane & 15;
  int bh = blockIdx.x;
  int b = bh >> 3, h = bh & 7;
  int q0 = blockIdx.y * 64 + wave * 16;          // this wave's 16 query rows

  const uint16_t* qbase = qb   + ((size_t)b * T_ * H_ + h) * D_;
  const uint16_t* kbase = kbuf + ((size_t)b * T_ * H_ + h) * D_;
  const uint16_t* vbase = vT   + ((size_t)(b * H_ + h)) * (size_t)D_ * T_;

  BFrag qf[4];                                   // Q rows, K=d split into 4x32
#pragma unroll
  for (int dk = 0; dk < 4; ++dk) {
    const uint16_t* qp = qbase + (size_t)(q0 + l16) * HD_ + dk * 32 + 8 * hl;
    qf[dk].q[0] = *(const u32x4*)qp;
    qf[dk].q[1] = *(const u32x4*)(qp + 16);
  }

  v8f o[8];
  float mrow[8], lrow[8];
#pragma unroll
  for (int nt = 0; nt < 8; ++nt) o[nt] = vzero8();
#pragma unroll
  for (int j = 0; j < 8; ++j) { mrow[j] = -1e30f; lrow[j] = 0.0f; }

  const float scale = 0.088388347648318447f;     // 1/sqrt(128)
  uint16_t* Pw = Pst[wave];
  int kend = q0 + 16;

  for (int kb0 = 0; kb0 < kend; kb0 += 32) {
    float pv[2][8], rowmax[8];
#pragma unroll
    for (int j = 0; j < 8; ++j) rowmax[j] = -1e30f;

#pragma unroll
    for (int s = 0; s < 2; ++s) {                // two 16-key sub-tiles
      int kt = kb0 + s * 16;
      v8f sc = vzero8();
#pragma unroll
      for (int dk = 0; dk < 4; ++dk) {
        BFrag kf;
        const uint16_t* kp = kbase + (size_t)(kt + l16) * HD_ + dk * 32 + 8 * hl;
        kf.q[0] = *(const u32x4*)kp;
        kf.q[1] = *(const u32x4*)(kp + 16);
        sc = __builtin_amdgcn_wmma_f32_16x16x32_bf16(
            false, qf[dk].v, false, kf.v, (short)0, sc, false, false);
      }
      int kidx = kt + l16;
#pragma unroll
      for (int j = 0; j < 8; ++j) {
        int qidx = q0 + 8 * hl + j;
        float v = sc[j] * scale;
        if (kidx > qidx) v = -1e30f;             // causal mask (also kills tiles past kend)
        pv[s][j] = v;
        rowmax[j] = fmaxf(rowmax[j], v);
      }
    }

    float alpha[8];
#pragma unroll
    for (int j = 0; j < 8; ++j) {
#pragma unroll
      for (int off = 1; off < 16; off <<= 1)     // reduce within 16-lane group (same rows)
        rowmax[j] = fmaxf(rowmax[j], __shfl_xor(rowmax[j], off, 32));
      float mnew = fmaxf(mrow[j], rowmax[j]);
      alpha[j] = __expf(mrow[j] - mnew);
      mrow[j] = mnew;
      float p0 = __expf(pv[0][j] - mnew);
      float p1 = __expf(pv[1][j] - mnew);
      pv[0][j] = p0; pv[1][j] = p1;
      float s = p0 + p1;
#pragma unroll
      for (int off = 1; off < 16; off <<= 1)
        s += __shfl_xor(s, off, 32);
      lrow[j] = lrow[j] * alpha[j] + s;
    }
#pragma unroll
    for (int nt = 0; nt < 8; ++nt)
#pragma unroll
      for (int j = 0; j < 8; ++j)
        o[nt][j] *= alpha[j];

    // C-layout probs -> LDS (row-major 16x32 bf16), per-wave region; LDS is in-order per wave
#pragma unroll
    for (int s = 0; s < 2; ++s)
#pragma unroll
      for (int j = 0; j < 8; ++j)
        Pw[(8 * hl + j) * 32 + s * 16 + l16] = f2bf(pv[s][j]);

    // reload as A fragment (16 q rows x 32 keys)
    BFrag pf;
    const uint16_t* lp = Pw + l16 * 32 + 8 * hl;
    pf.q[0] = *(const u32x4*)lp;
    pf.q[1] = *(const u32x4*)(lp + 16);

    // accumulate P @ V over 8 d-tiles; V^T[d][t] gives contiguous-K fragments
#pragma unroll
    for (int nt = 0; nt < 8; ++nt) {
      BFrag vf;
      const uint16_t* vp = vbase + (size_t)(nt * 16 + l16) * T_ + kb0 + 8 * hl;
      vf.q[0] = *(const u32x4*)vp;
      vf.q[1] = *(const u32x4*)(vp + 16);
      o[nt] = __builtin_amdgcn_wmma_f32_16x16x32_bf16(
          false, pf.v, false, vf.v, (short)0, o[nt], false, false);
    }
  }

#pragma unroll
  for (int j = 0; j < 8; ++j) {
    float inv = 1.0f / lrow[j];
    int t = q0 + 8 * hl + j;
    uint16_t* op = attnb + (((size_t)b * T_ + t) * H_ + h) * D_;
#pragma unroll
    for (int nt = 0; nt < 8; ++nt)
      op[nt * 16 + l16] = f2bf(o[nt][j] * inv);
  }
}

// ---------------- residual add + layernorm (one token per block) ----------------
__global__ __launch_bounds__(128) void addln_kernel(
    const float* __restrict__ x, float* __restrict__ hid, uint16_t* __restrict__ hidb,
    const float* __restrict__ g, const float* __restrict__ bb)
{
  __shared__ float red1[4], red2[4];
  int tok = blockIdx.x, i = threadIdx.x;
  int lane = i & 31, wave = i >> 5;
  size_t base = (size_t)tok * D_;
  float v = x[base + i] + hid[base + i];
  float s = v;
#pragma unroll
  for (int off = 1; off < 32; off <<= 1) s += __shfl_xor(s, off, 32);
  if (lane == 0) red1[wave] = s;
  __syncthreads();
  float mu = (red1[0] + red1[1] + red1[2] + red1[3]) * (1.0f / D_);
  float d = v - mu;
  float s2 = d * d;
#pragma unroll
  for (int off = 1; off < 32; off <<= 1) s2 += __shfl_xor(s2, off, 32);
  if (lane == 0) red2[wave] = s2;
  __syncthreads();
  float var = (red2[0] + red2[1] + red2[2] + red2[3]) * (1.0f / D_);
  float y = d * rsqrtf(var + 1e-5f) * g[i] + bb[i];
  hid[base + i]  = y;
  hidb[base + i] = f2bf(y);
}

// ---------------- mean pool over T ----------------
__global__ __launch_bounds__(256) void pool_kernel(
    const float* __restrict__ hid, float* __restrict__ pooled)
{
  int i = threadIdx.x;                 // 256 = B_*D_
  int b = i >> 7, d = i & 127;
  const float* hp = hid + (size_t)b * T_ * D_ + d;
  float s = 0.0f;
  for (int t = 0; t < T_; ++t) s += hp[(size_t)t * D_];
  pooled[i] = s * (1.0f / T_);
}

// ---------------- logits = pooled @ out_w + out_b ----------------
__global__ __launch_bounds__(128) void logits_kernel(
    const float* __restrict__ pooled, const float* __restrict__ W,
    const float* __restrict__ bias, float* __restrict__ logits)
{
  int c = blockIdx.x * 128 + threadIdx.x;
  float s0 = bias[c], s1 = bias[c];
  for (int d = 0; d < D_; ++d) {
    float w = W[(size_t)d * C_ + c];
    s0 += pooled[d] * w;
    s1 += pooled[D_ + d] * w;
  }
  logits[c]      = s0;
  logits[C_ + c] = s1;
}

// ---------------- log_softmax over classes ----------------
__global__ __launch_bounds__(256) void lsm_kernel(
    const float* __restrict__ logits, float* __restrict__ out)
{
  __shared__ float red1[8], red2[8];
  int b = blockIdx.x, tid = threadIdx.x;
  int lane = tid & 31, wave = tid >> 5;
  const float* lp = logits + (size_t)b * C_;
  float mx = -1e30f;
  for (int c = tid; c < C_; c += 256) mx = fmaxf(mx, lp[c]);
#pragma unroll
  for (int off = 1; off < 32; off <<= 1) mx = fmaxf(mx, __shfl_xor(mx, off, 32));
  if (lane == 0) red1[wave] = mx;
  __syncthreads();
  mx = red1[0];
#pragma unroll
  for (int w = 1; w < 8; ++w) mx = fmaxf(mx, red1[w]);
  float se = 0.0f;
  for (int c = tid; c < C_; c += 256) se += __expf(lp[c] - mx);
#pragma unroll
  for (int off = 1; off < 32; off <<= 1) se += __shfl_xor(se, off, 32);
  if (lane == 0) red2[wave] = se;
  __syncthreads();
  se = red2[0] + red2[1] + red2[2] + red2[3] + red2[4] + red2[5] + red2[6] + red2[7];
  float lse = mx + __logf(se);
  for (int c = tid; c < C_; c += 256) out[(size_t)b * C_ + c] = lp[c] - lse;
}

// =======================================================================
extern "C" void kernel_launch(void* const* d_in, const int* in_sizes, int n_in,
                              void* d_out, int out_size, void* d_ws, size_t ws_size,
                              hipStream_t stream)
{
  (void)in_sizes; (void)n_in; (void)out_size; (void)ws_size;
  const int*   x      = (const int*)  d_in[0];
  const float* tokemb = (const float*)d_in[1];
  const float* posemb = (const float*)d_in[2];
  const float* Wq     = (const float*)d_in[3];
  const float* Wk     = (const float*)d_in[4];
  const float* Wv     = (const float*)d_in[5];
  const float* Wu     = (const float*)d_in[6];
  const float* bu     = (const float*)d_in[7];
  const float* ln1g   = (const float*)d_in[8];
  const float* ln1b   = (const float*)d_in[9];
  const float* fw1    = (const float*)d_in[10];
  const float* fb1    = (const float*)d_in[11];
  const float* fw2    = (const float*)d_in[12];
  const float* fb2    = (const float*)d_in[13];
  const float* ln2g   = (const float*)d_in[14];
  const float* ln2b   = (const float*)d_in[15];
  const float* outw   = (const float*)d_in[16];
  const float* outb   = (const float*)d_in[17];

  char* ws = (char*)d_ws;
  size_t off = 0;
  auto alloc = [&](size_t bytes) -> char* {
    off = (off + 255) & ~(size_t)255;
    char* p = ws + off;
    off += bytes;
    return p;
  };

  float*    hid    = (float*)   alloc((size_t)BT_ * D_ * 4);
  uint16_t* hidb   = (uint16_t*)alloc((size_t)BT_ * D_ * 2);
  uint16_t* qbuf   = (uint16_t*)alloc((size_t)BT_ * HD_ * 2);
  uint16_t* kbuf   = (uint16_t*)alloc((size_t)BT_ * HD_ * 2);
  uint16_t* vbuf   = (uint16_t*)alloc((size_t)BT_ * HD_ * 2);
  uint16_t* vTbuf  = (uint16_t*)alloc((size_t)BT_ * HD_ * 2);
  uint16_t* attnb  = (uint16_t*)alloc((size_t)BT_ * HD_ * 2);
  float*    proj   = (float*)   alloc((size_t)BT_ * D_ * 4);
  uint16_t* h1b    = (uint16_t*)alloc((size_t)BT_ * 4 * D_ * 2);
  float*    h2     = (float*)   alloc((size_t)BT_ * D_ * 4);
  uint16_t* WqT    = (uint16_t*)alloc((size_t)D_ * HD_ * 2);
  uint16_t* WkT    = (uint16_t*)alloc((size_t)D_ * HD_ * 2);
  uint16_t* WvT    = (uint16_t*)alloc((size_t)D_ * HD_ * 2);
  uint16_t* WuT    = (uint16_t*)alloc((size_t)HD_ * D_ * 2);
  uint16_t* f1T    = (uint16_t*)alloc((size_t)D_ * 4 * D_ * 2);
  uint16_t* f2T    = (uint16_t*)alloc((size_t)4 * D_ * D_ * 2);
  float*    pooled = (float*)   alloc((size_t)B_ * D_ * 4);
  float*    lgts   = (float*)   alloc((size_t)B_ * C_ * 4);

  embed_kernel<<<BT_, 128, 0, stream>>>(x, tokemb, posemb, hid, hidb);

  for (int L = 0; L < DEPTH_; ++L) {
    transw_kernel<<<512, 256, 0, stream>>>(Wq + (size_t)L * D_ * HD_, WqT, D_, HD_);
    transw_kernel<<<512, 256, 0, stream>>>(Wk + (size_t)L * D_ * HD_, WkT, D_, HD_);
    transw_kernel<<<512, 256, 0, stream>>>(Wv + (size_t)L * D_ * HD_, WvT, D_, HD_);
    transw_kernel<<<512, 256, 0, stream>>>(Wu + (size_t)L * HD_ * D_, WuT, HD_, D_);
    transw_kernel<<<256, 256, 0, stream>>>(fw1 + (size_t)L * D_ * 4 * D_, f1T, D_, 4 * D_);
    transw_kernel<<<256, 256, 0, stream>>>(fw2 + (size_t)L * 4 * D_ * D_, f2T, 4 * D_, D_);

    gemm_kernel<<<dim3(16, 64), 128, 0, stream>>>(hidb, WqT, nullptr, nullptr, qbuf, BT_, HD_, D_, 0);
    gemm_kernel<<<dim3(16, 64), 128, 0, stream>>>(hidb, WkT, nullptr, nullptr, kbuf, BT_, HD_, D_, 0);
    gemm_kernel<<<dim3(16, 64), 128, 0, stream>>>(hidb, WvT, nullptr, nullptr, vbuf, BT_, HD_, D_, 0);
    transv_kernel<<<(B_ * H_ * D_ * T_) / 256, 256, 0, stream>>>(vbuf, vTbuf);
    attn_kernel<<<dim3(B_ * H_, T_ / 64), 128, 0, stream>>>(qbuf, kbuf, vTbuf, attnb);
    gemm_kernel<<<dim3(2, 64), 128, 0, stream>>>(attnb, WuT, bu + (size_t)L * D_, proj, nullptr, BT_, D_, HD_, 0);
    addln_kernel<<<BT_, 128, 0, stream>>>(proj, hid, hidb, ln1g + (size_t)L * D_, ln1b + (size_t)L * D_);
    gemm_kernel<<<dim3(8, 64), 128, 0, stream>>>(hidb, f1T, fb1 + (size_t)L * 4 * D_, nullptr, h1b, BT_, 4 * D_, D_, 1);
    gemm_kernel<<<dim3(2, 64), 128, 0, stream>>>(h1b, f2T, fb2 + (size_t)L * D_, h2, nullptr, BT_, D_, 4 * D_, 0);
    addln_kernel<<<BT_, 128, 0, stream>>>(h2, hid, hidb, ln2g + (size_t)L * D_, ln2b + (size_t)L * D_);
  }

  pool_kernel<<<1, 256, 0, stream>>>(hid, pooled);
  logits_kernel<<<C_ / 128, 128, 0, stream>>>(pooled, outw, outb, lgts);
  lsm_kernel<<<B_, 256, 0, stream>>>(lgts, (float*)d_out);
}
